// TransformerBlock_90898687852797
// MI455X (gfx1250) — compile-verified, matrix operand order checked by reading, not run on hardware
//
#include <hip/hip_runtime.h>
#include <hip/hip_bf16.h>
#include <cmath>

typedef __attribute__((ext_vector_type(16))) _Float16 v16h;
typedef __attribute__((ext_vector_type(8)))  _Float16 v8h;
typedef __attribute__((ext_vector_type(8)))  float    v8f;

#define NTOK 25600
#define CCH  96
#define NHD  4
#define HDD  24
#define LLOC 27
#define PLP  50
#define TT   4096
// conv volume (X=40, Y=40, Z=16); padded (42,42,18)
#define PSTRC 31752   // 42*42*18
#define PSTRX 756     // 42*18
#define PSTRY 18
#define KC    2592    // 27*96, tap-major: k = t*96 + ci

__device__ __forceinline__ v8f wmma32(v16h a, v16h b, v8f c) {
  return __builtin_amdgcn_wmma_f32_16x16x32_f16(false, a, false, b, (short)0, c, false, false);
}

// Row-major f16 fragment load: rows r0..r0+15, cols k0..k0+31
// lane&15 -> row; (lane>>4)*8 -> which K-half (per 16-bit A/B VGPR layout)
__device__ __forceinline__ v16h frag_ld(const _Float16* base, int ld, int r0, int k0, int lane) {
  const _Float16* p = base + (r0 + (lane & 15)) * ld + k0 + ((lane >> 4) << 3);
  v16h f;
#pragma unroll
  for (int i = 0; i < 8; ++i) f[i] = p[i];
#pragma unroll
  for (int i = 0; i < 8; ++i) f[8 + i] = p[16 + i];
  return f;
}

// ---- CDNA5 async global->LDS (ASYNCcnt) ----
__device__ __forceinline__ void async_b128(unsigned lds_off, const _Float16* g) {
  asm volatile("global_load_async_to_lds_b128 %0, %1, off"
               :: "v"(lds_off), "v"(g) : "memory");
}
__device__ __forceinline__ void wait_async0() {
  asm volatile("s_wait_asynccnt 0x0" ::: "memory");
}

// ---------------- f32 -> f16 convert ----------------
__global__ void k_cvt(const float* __restrict__ s, _Float16* __restrict__ d, int n) {
  int i = blockIdx.x * blockDim.x + threadIdx.x;
  if (i < n) d[i] = (_Float16)s[i];
}

// ---------------- conv weight convert + permute to tap-major K ----------------
__global__ void k_cvtw(const float* __restrict__ s, _Float16* __restrict__ d) {
  int i = blockIdx.x * blockDim.x + threadIdx.x;  // over 96*2592 (dst order)
  if (i >= CCH * KC) return;
  int co = i / KC, k = i % KC;                    // dst k = t*96 + ci
  int t = k / CCH, ci = k % CCH;
  d[i] = (_Float16)s[co * KC + ci * 27 + t];      // src k = ci*27 + t
}

// ---------------- xt = x^T + 1 ; layernorm -> f16 ----------------
__global__ void k_ln(const float* __restrict__ x, const float* __restrict__ w,
                     const float* __restrict__ b, float* __restrict__ xt,
                     _Float16* __restrict__ xnh) {
  int n = blockIdx.x * blockDim.x + threadIdx.x;
  if (n >= NTOK) return;
  float s = 0.f, ss = 0.f;
  for (int c = 0; c < CCH; ++c) { float v = x[c * NTOK + n] + 1.f; s += v; ss += v * v; }
  float mu = s / CCH;
  float inv = rsqrtf(ss / CCH - mu * mu + 1e-5f);
  for (int c = 0; c < CCH; ++c) {
    float v = x[c * NTOK + n] + 1.f;
    xt[n * CCH + c] = v;
    xnh[n * CCH + c] = (_Float16)((v - mu) * inv * w[c] + b[c]);
  }
}

// -------- generic WMMA GEMM (K == 96): C[m, colofs+n] = A[m,:]·W[n,:] + bias, opt GELU --------
__global__ void k_gemm(const _Float16* __restrict__ A, const _Float16* __restrict__ W,
                       float* __restrict__ Cout, int Nout,
                       const float* __restrict__ bias, int act, int ld, int colofs) {
  int lane = threadIdx.x & 31;
  int wid  = threadIdx.x >> 5;
  int m0   = blockIdx.x << 4;
  v16h afr[3];
#pragma unroll
  for (int kk = 0; kk < 3; ++kk) afr[kk] = frag_ld(A, CCH, m0, kk * 32, lane);
  int ntiles = Nout >> 4;
  for (int nt = wid; nt < ntiles; nt += 8) {
    int n0 = nt << 4;
    v8f acc = {};
#pragma unroll
    for (int kk = 0; kk < 3; ++kk) {
      v16h b = frag_ld(W, CCH, n0, kk * 32, lane);
      acc = wmma32(afr[kk], b, acc);
    }
    int col = n0 + (lane & 15);
    float bv = bias ? bias[col] : 0.f;
    int mrow = m0 + ((lane >> 4) << 3);
#pragma unroll
    for (int r = 0; r < 8; ++r) {
      float v = acc[r] + bv;
      if (act == 1) v = 0.5f * v * (1.f + erff(v * 0.70710678f));  // exact GELU
      Cout[(mrow + r) * ld + colofs + col] = v;
    }
  }
}

// ---------------- 8x8x8 average pool of gelu(sr(xn)) ----------------
__global__ void k_pool_avg(const float* __restrict__ ysr, float* __restrict__ xpp) {
  __shared__ float red[256];
  int p = blockIdx.x / CCH, c = blockIdx.x % CCH;
  int pd = p % 2, pw = (p / 2) % 5, ph = p / 10;
  float s = 0.f;
  for (int e = threadIdx.x; e < 512; e += 256) {
    int a  = ph * 8 + (e >> 6);
    int bb = pw * 8 + ((e >> 3) & 7);
    int cz = pd * 8 + (e & 7);
    s += ysr[(a * 640 + bb * 16 + cz) * CCH + c];
  }
  red[threadIdx.x] = s;
  __syncthreads();
  for (int off = 128; off > 0; off >>= 1) {
    if (threadIdx.x < off) red[threadIdx.x] += red[threadIdx.x + off];
    __syncthreads();
  }
  if (threadIdx.x == 0) xpp[p * CCH + c] = red[0] * (1.f / 512.f);
}

// ---------------- pool LN + kv GEMM + l2(k_pool) (tiny: PL=50) ----------------
__global__ void k_pool_lnkv(const float* __restrict__ xpp, const float* __restrict__ lw,
                            const float* __restrict__ lb, const float* __restrict__ wkv,
                            float* __restrict__ kpn, float* __restrict__ vpool) {
  int p = threadIdx.x;
  if (p >= PLP) return;
  float row[CCH], kvp[2 * CCH];
  float s = 0.f, ss = 0.f;
  for (int c = 0; c < CCH; ++c) { float v = xpp[p * CCH + c]; s += v; ss += v * v; }
  float mu = s / CCH, inv = rsqrtf(ss / CCH - mu * mu + 1e-5f);
  for (int c = 0; c < CCH; ++c) row[c] = (xpp[p * CCH + c] - mu) * inv * lw[c] + lb[c];
  for (int j = 0; j < 2 * CCH; ++j) {
    float a = 0.f;
    for (int c = 0; c < CCH; ++c) a += row[c] * wkv[j * CCH + c];
    kvp[j] = a;
  }
  for (int h = 0; h < NHD; ++h) {
    float nrm = 0.f;
    for (int d = 0; d < HDD; ++d) { float v = kvp[h * HDD + d]; nrm += v * v; }
    nrm = fmaxf(sqrtf(nrm), 1e-12f);
    for (int d = 0; d < HDD; ++d) {
      kpn  [(h * PLP + p) * HDD + d] = kvp[h * HDD + d] / nrm;
      vpool[(h * PLP + p) * HDD + d] = kvp[CCH + h * HDD + d];
    }
  }
}

// ---------------- CPB MLP table: tab[t][h] ----------------
__global__ void k_cpb(const float* __restrict__ rct, const float* __restrict__ w1,
                      const float* __restrict__ b1, const float* __restrict__ w2,
                      const float* __restrict__ b2, float* __restrict__ tab) {
  int t = blockIdx.x * blockDim.x + threadIdx.x;
  if (t >= TT) return;
  float r0 = rct[t * 3], r1 = rct[t * 3 + 1], r2 = rct[t * 3 + 2];
  float a0 = 0.f, a1 = 0.f, a2 = 0.f, a3 = 0.f;
  for (int j = 0; j < 512; ++j) {
    float hv = fmaxf(r0 * w1[j * 3] + r1 * w1[j * 3 + 1] + r2 * w1[j * 3 + 2] + b1[j], 0.f);
    a0 += hv * w2[j]; a1 += hv * w2[512 + j]; a2 += hv * w2[1024 + j]; a3 += hv * w2[1536 + j];
  }
  tab[t * 4 + 0] = a0 + b2[0]; tab[t * 4 + 1] = a1 + b2[1];
  tab[t * 4 + 2] = a2 + b2[2]; tab[t * 4 + 3] = a3 + b2[3];
}

// ---------------- local+pool attention: block=token, wave=head ----------------
__global__ void k_attn(const float* __restrict__ Y, const float* __restrict__ qe,
                       const float* __restrict__ temp, const float* __restrict__ sls,
                       const float* __restrict__ rbl, const float* __restrict__ lt,
                       const float* __restrict__ lbias, const float* __restrict__ kpn,
                       const float* __restrict__ vpool, const float* __restrict__ tab,
                       const int* __restrict__ rpi, _Float16* __restrict__ xsa) {
  __shared__ float lqs[NHD][HDD], lqn[NHD][HDD], la[NHD][80];
  int n = blockIdx.x;
  int h = threadIdx.x >> 5;
  int lane = threadIdx.x & 31;
  const float* q = Y + n * 384 + h * HDD;
  float qd = (lane < HDD) ? q[lane] : 0.f;
  float sq = qd * qd;
  for (int m = 16; m; m >>= 1) sq += __shfl_xor(sq, m, 32);
  float nrm = fmaxf(sqrtf(sq), 1e-12f);
  float tv = temp[h];
  float tscale = (tv > 20.f ? tv : log1pf(expf(tv))) * sls[0];
  if (lane < HDD) {
    float qn = qd / nrm;
    lqn[h][lane] = qn;
    lqs[h][lane] = (qn + qe[h * HDD + lane]) * tscale;
  }
  __syncthreads();
  int dz = n / 1600, rem = n % 1600, hy = rem / 40, wx = rem % 40;
  float sc[3];
#pragma unroll
  for (int ii = 0; ii < 3; ++ii) {
    int s = lane + ii * 32;
    sc[ii] = -INFINITY;
    if (s < LLOC) {
      int nz = dz + s / 9 - 1, ny = hy + (s / 3) % 3 - 1, nx = wx + s % 3 - 1;
      if (nz >= 0 && nz < 16 && ny >= 0 && ny < 40 && nx >= 0 && nx < 40) {
        const float* kk = Y + ((nz * 40 + ny) * 40 + nx) * 384 + 96 + h * HDD;
        float a = 0.f;
        for (int d = 0; d < HDD; ++d) a += lqs[h][d] * kk[d];
        sc[ii] = a + rbl[h * LLOC + s];
      }
    } else if (s < LLOC + PLP) {
      int p = s - LLOC;
      const float* kk = kpn + (h * PLP + p) * HDD;
      float a = 0.f;
      for (int d = 0; d < HDD; ++d) a += lqs[h][d] * kk[d];
      sc[ii] = a + tab[rpi[n * PLP + p] * NHD + h];
    }
  }
  float mx = fmaxf(fmaxf(sc[0], sc[1]), sc[2]);
  for (int m = 16; m; m >>= 1) mx = fmaxf(mx, __shfl_xor(mx, m, 32));
  float es[3], sum = 0.f;
#pragma unroll
  for (int ii = 0; ii < 3; ++ii) { es[ii] = (sc[ii] == -INFINITY) ? 0.f : expf(sc[ii] - mx); sum += es[ii]; }
  for (int m = 16; m; m >>= 1) sum += __shfl_xor(sum, m, 32);
  float rinv = 1.f / sum;
#pragma unroll
  for (int ii = 0; ii < 3; ++ii) {
    int s = lane + ii * 32;
    if (s < LLOC + PLP) {
      float a = es[ii] * rinv;
      if (s < LLOC) {  // add learnable-token term (unmasked, matches reference)
        float add = lbias[h * LLOC + s];
        for (int d = 0; d < HDD; ++d) add += lqn[h][d] * lt[(h * HDD + d) * LLOC + s];
        a += add;
      }
      la[h][s] = a;
    }
  }
  __syncthreads();
  if (lane < HDD) {
    float acc = 0.f;
    for (int t = 0; t < LLOC; ++t) {
      int nz = dz + t / 9 - 1, ny = hy + (t / 3) % 3 - 1, nx = wx + t % 3 - 1;
      if (nz >= 0 && nz < 16 && ny >= 0 && ny < 40 && nx >= 0 && nx < 40)
        acc += la[h][t] * Y[((nz * 40 + ny) * 40 + nx) * 384 + 288 + h * HDD + lane];
    }
    for (int p = 0; p < PLP; ++p)
      acc += la[h][LLOC + p] * vpool[(h * PLP + p) * HDD + lane];
    xsa[n * CCH + h * HDD + lane] = (_Float16)acc;
  }
}

// ---------------- column L2 norms of q (cols 0..95) / k (96..191) over N ----------------
__global__ void k_colnorm(const float* __restrict__ Y, float* __restrict__ norms) {
  __shared__ float red[256];
  int j = blockIdx.x;
  float s = 0.f;
  for (int n = threadIdx.x; n < NTOK; n += 256) { float v = Y[n * 384 + j]; s += v * v; }
  red[threadIdx.x] = s;
  __syncthreads();
  for (int off = 128; off > 0; off >>= 1) {
    if (threadIdx.x < off) red[threadIdx.x] += red[threadIdx.x + off];
    __syncthreads();
  }
  if (threadIdx.x == 0) norms[j] = fmaxf(sqrtf(red[0]), 1e-12f);
}

// ---------------- channel-attn Gram via WMMA: G[h][d][e] = sum_n q*k ; 32-way K-split ----------------
__global__ void k_gram(const float* __restrict__ Y, float* __restrict__ gram) {
  int lane  = threadIdx.x & 31;
  int h     = blockIdx.x >> 5;
  int split = blockIdx.x & 31;
  v8f acc[2][2] = {};
  int kt0 = split * 25;  // 800 k-tiles of 32 over N=25600
  for (int kt = kt0; kt < kt0 + 25; ++kt) {
    int kbase = kt * 32 + ((lane >> 4) << 3);
    v16h a[2], b[2];
#pragma unroll
    for (int mt = 0; mt < 2; ++mt) {
      int d = mt * 16 + (lane & 15);
#pragma unroll
      for (int i = 0; i < 8; ++i) {
        a[mt][i]     = (d < HDD) ? (_Float16)Y[(kbase + i) * 384 + h * HDD + d]      : (_Float16)0.f;
        a[mt][8 + i] = (d < HDD) ? (_Float16)Y[(kbase + 16 + i) * 384 + h * HDD + d] : (_Float16)0.f;
      }
    }
#pragma unroll
    for (int nt = 0; nt < 2; ++nt) {
      int e = nt * 16 + (lane & 15);
#pragma unroll
      for (int i = 0; i < 8; ++i) {
        b[nt][i]     = (e < HDD) ? (_Float16)Y[(kbase + i) * 384 + 96 + h * HDD + e]      : (_Float16)0.f;
        b[nt][8 + i] = (e < HDD) ? (_Float16)Y[(kbase + 16 + i) * 384 + 96 + h * HDD + e] : (_Float16)0.f;
      }
    }
#pragma unroll
    for (int mt = 0; mt < 2; ++mt)
#pragma unroll
      for (int nt = 0; nt < 2; ++nt)
        acc[mt][nt] = wmma32(a[mt], b[nt], acc[mt][nt]);
  }
#pragma unroll
  for (int mt = 0; mt < 2; ++mt)
#pragma unroll
    for (int nt = 0; nt < 2; ++nt) {
      int e = nt * 16 + (lane & 15);
      if (e >= HDD) continue;
      int dbase = mt * 16 + ((lane >> 4) << 3);
#pragma unroll
      for (int r = 0; r < 8; ++r) {
        int d = dbase + r;
        if (d < HDD) atomicAdd(&gram[(h * HDD + d) * HDD + e], acc[mt][nt][r]);
      }
    }
}

// ---------------- channel-attn softmax rows ----------------
__global__ void k_casoft(const float* __restrict__ gram, const float* __restrict__ norms,
                         const float* __restrict__ t2, float* __restrict__ A) {
  int r = threadIdx.x;
  if (r >= 96) return;
  int h = r / HDD, d = r % HDD;
  float row[HDD], mx = -INFINITY;
  for (int e = 0; e < HDD; ++e) {
    float v = gram[(h * HDD + d) * HDD + e] / (norms[h * HDD + d] * norms[96 + h * HDD + e]) * t2[h];
    row[e] = v; mx = fmaxf(mx, v);
  }
  float s = 0.f;
  for (int e = 0; e < HDD; ++e) { row[e] = expf(row[e] - mx); s += row[e]; }
  float inv = 1.f / s;
  for (int e = 0; e < HDD; ++e) A[(h * HDD + d) * HDD + e] = row[e] * inv;
}

// ---------------- x_ca = A_ca @ v_ca ----------------
__global__ void k_xca(const float* __restrict__ Y, const float* __restrict__ A,
                      _Float16* __restrict__ xca) {
  __shared__ float As[NHD * HDD * HDD];
  for (int i = threadIdx.x; i < NHD * HDD * HDD; i += 256) As[i] = A[i];
  __syncthreads();
  int idx = blockIdx.x * 256 + threadIdx.x;
  if (idx >= NTOK * CCH) return;
  int n = idx / CCH, c = idx % CCH;
  int h = c / HDD, d = c % HDD;
  const float* v = Y + n * 384 + 192 + h * HDD;
  float acc = 0.f;
  for (int e = 0; e < HDD; ++e) acc += As[(h * HDD + d) * HDD + e] * v[e];
  xca[idx] = (_Float16)acc;
}

// ---------------- ax = xt + gamma*epa -> skip_f32 (C-major) + padded f16 ----------------
__global__ void k_ax(const float* __restrict__ xt, const float* __restrict__ gamma,
                     const float* __restrict__ epa, float* __restrict__ skip,
                     _Float16* __restrict__ padS) {
  int idx = blockIdx.x * 256 + threadIdx.x;
  if (idx >= NTOK * CCH) return;
  int n = idx / CCH, c = idx % CCH;
  float v = xt[idx] + gamma[c] * epa[idx];
  skip[c * NTOK + n] = v;
  int a = n / 640, bb = (n / 16) % 40, cz = n % 16;
  padS[c * PSTRC + (a + 1) * PSTRX + (bb + 1) * PSTRY + (cz + 1)] = (_Float16)v;
}

// ---- 3x3x3 conv as implicit WMMA GEMM (tap-major K), async-LDS double-buffered weights ----
__global__ void k_conv3(const _Float16* __restrict__ pin, const _Float16* __restrict__ wgt,
                        const float* __restrict__ bnp, const float* __restrict__ res,
                        _Float16* __restrict__ pout, float* __restrict__ f32out) {
  __shared__ __attribute__((aligned(16))) _Float16 wslab[2][CCH * 32];
  int lane = threadIdx.x & 31;
  int wid  = threadIdx.x >> 5;
  int tile = blockIdx.x * 8 + wid;   // one (x,y) column of 16 z positions per wave
  int a = tile / 40, bb = tile % 40;
  int z = lane & 15;
  int chalf = (lane >> 4) << 3;
  const _Float16* base = pin + (a + 1) * PSTRX + (bb + 1) * PSTRY + (z + 1);

  // stage weight k-tile 0 (96 co x 32 k halves = 384 b128 pieces)
  for (int idx = threadIdx.x; idx < 384; idx += 256) {
    int co = idx >> 2, part = idx & 3;
    async_b128((unsigned)(uintptr_t)&wslab[0][co * 32 + part * 8],
               wgt + co * KC + part * 8);
  }

  v8f acc[6] = {};
  for (int kt = 0; kt < 81; ++kt) {
    int buf = kt & 1;
    wait_async0();      // my async copies (tile kt) done
    __syncthreads();    // everyone's copies done; previous compute done
    if (kt < 80) {      // prefetch next weight tile into other buffer
      int kb2 = (kt + 1) * 32;
      for (int idx = threadIdx.x; idx < 384; idx += 256) {
        int co = idx >> 2, part = idx & 3;
        async_b128((unsigned)(uintptr_t)&wslab[1 - buf][co * 32 + part * 8],
                   wgt + co * KC + kb2 + part * 8);
      }
    }
    // A fragment: tap t = kt/3 constant over the tile; channels ci = (kt%3)*32 + pattern
    int t = kt / 3;
    const _Float16* ab = base + (t / 9 - 1) * PSTRX + ((t / 3) % 3 - 1) * PSTRY + (t % 3 - 1)
                       + ((kt % 3) * 32 + chalf) * PSTRC;
    v16h af;
#pragma unroll
    for (int i = 0; i < 8; ++i) af[i] = ab[i * PSTRC];
#pragma unroll
    for (int i = 0; i < 8; ++i) af[8 + i] = ab[(16 + i) * PSTRC];
#pragma unroll
    for (int ct = 0; ct < 6; ++ct) {
      int co = ct * 16 + (lane & 15);
      v8h b0 = *(const v8h*)&wslab[buf][co * 32 + chalf];
      v8h b1 = *(const v8h*)&wslab[buf][co * 32 + chalf + 16];
      v16h bf = __builtin_shufflevector(b0, b1, 0, 1, 2, 3, 4, 5, 6, 7,
                                        8, 9, 10, 11, 12, 13, 14, 15);
      acc[ct] = wmma32(af, bf, acc[ct]);
    }
  }
#pragma unroll
  for (int ct = 0; ct < 6; ++ct) {
    int co = ct * 16 + (lane & 15);
    float g = bnp[co], be = bnp[96 + co], mu = bnp[192 + co], va = bnp[288 + co];
    float s = g * rsqrtf(va + 1e-5f);
    float sh = be - mu * s;
    int zb = (lane >> 4) << 3;
#pragma unroll
    for (int r = 0; r < 8; ++r) {
      int zz = zb + r;
      int p = (a * 40 + bb) * 16 + zz;
      float v = acc[ct][r] * s + sh;
      if (res) v += res[co * NTOK + p];
      v = (v >= 0.f) ? v : 0.01f * v;
      pout[co * PSTRC + (a + 1) * PSTRX + (bb + 1) * PSTRY + (zz + 1)] = (_Float16)v;
      if (f32out) f32out[co * NTOK + p] = v;
    }
  }
}

// ---------------- out = skip + conv8(h4) + b : WMMA, M=co, N=positions ----------------
__global__ void k_final(const _Float16* __restrict__ h4, const _Float16* __restrict__ w8,
                        const float* __restrict__ skip, const float* __restrict__ b8,
                        float* __restrict__ out) {
  int lane = threadIdx.x & 31;
  int wid  = threadIdx.x >> 5;
  int tile = blockIdx.x * 8 + wid;
  int a = tile / 40, bb = tile % 40;
  int z = lane & 15;
  const _Float16* base = h4 + (a + 1) * PSTRX + (bb + 1) * PSTRY + (z + 1);
  int chalf = (lane >> 4) << 3;
  for (int ot = 0; ot < 6; ++ot) {
    v8f acc = {};
    for (int kt = 0; kt < 3; ++kt) {
      v16h af = frag_ld(w8, CCH, ot * 16, kt * 32, lane);  // A = conv8_w [o][c]
      int kb = kt * 32 + chalf;
      v16h bf;
#pragma unroll
      for (int i = 0; i < 8; ++i) bf[i] = base[(kb + i) * PSTRC];
#pragma unroll
      for (int i = 0; i < 8; ++i) bf[8 + i] = base[(kb + 16 + i) * PSTRC];
      acc = wmma32(af, bf, acc);
    }
    int p = (a * 40 + bb) * 16 + z;
    int ob = ot * 16 + ((lane >> 4) << 3);
#pragma unroll
    for (int r = 0; r < 8; ++r) {
      int o = ob + r;
      out[o * NTOK + p] = skip[o * NTOK + p] + acc[r] + b8[o];
    }
  }
}

extern "C" void kernel_launch(void* const* d_in, const int* in_sizes, int n_in,
                              void* d_out, int out_size, void* d_ws, size_t ws_size,
                              hipStream_t stream) {
  (void)in_sizes; (void)n_in; (void)out_size; (void)ws_size;
  const float* x      = (const float*)d_in[0];
  const int*   rpi    = (const int*)  d_in[1];
  const float* rct    = (const float*)d_in[2];
  const float* sls    = (const float*)d_in[3];
  const float* norm_w = (const float*)d_in[4];
  const float* norm_b = (const float*)d_in[5];
  const float* gamma  = (const float*)d_in[6];
  const float* w_qkvv = (const float*)d_in[7];
  const float* temp   = (const float*)d_in[8];
  const float* temp2  = (const float*)d_in[9];
  const float* qe     = (const float*)d_in[10];
  const float* rbl    = (const float*)d_in[11];
  const float* lt     = (const float*)d_in[12];
  const float* lb     = (const float*)d_in[13];
  const float* sr_w   = (const float*)d_in[14];
  const float* sr_b   = (const float*)d_in[15];
  const float* enw    = (const float*)d_in[16];
  const float* enb    = (const float*)d_in[17];
  const float* w_kv   = (const float*)d_in[18];
  const float* cpb1w  = (const float*)d_in[19];
  const float* cpb1b  = (const float*)d_in[20];
  const float* cpb2w  = (const float*)d_in[21];
  const float* cpb2b  = (const float*)d_in[22];
  const float* o1w    = (const float*)d_in[23];
  const float* o1b    = (const float*)d_in[24];
  const float* o2w    = (const float*)d_in[25];
  const float* o2b    = (const float*)d_in[26];
  const float* c51aw  = (const float*)d_in[27];
  const float* c51abn = (const float*)d_in[28];
  const float* c51bw  = (const float*)d_in[29];
  const float* c51bbn = (const float*)d_in[30];
  const float* c52aw  = (const float*)d_in[31];
  const float* c52abn = (const float*)d_in[32];
  const float* c52bw  = (const float*)d_in[33];
  const float* c52bbn = (const float*)d_in[34];
  const float* c8w    = (const float*)d_in[35];
  const float* c8b    = (const float*)d_in[36];

  char* ws = (char*)d_ws;
  size_t off = 0;
  auto alloc = [&](size_t bytes) { size_t o = off; off += (bytes + 255) & ~(size_t)255; return ws + o; };

  float*    XT    = (float*)   alloc((size_t)NTOK * CCH * 4);
  _Float16* XNH   = (_Float16*)alloc((size_t)NTOK * CCH * 2);
  float*    Y     = (float*)   alloc((size_t)NTOK * 384 * 4);
  float*    YSR   = (float*)   alloc((size_t)NTOK * CCH * 4);
  float*    XPP   = (float*)   alloc((size_t)PLP * CCH * 4);
  float*    KPN   = (float*)   alloc((size_t)NHD * PLP * HDD * 4);
  float*    VPOOL = (float*)   alloc((size_t)NHD * PLP * HDD * 4);
  float*    TAB   = (float*)   alloc((size_t)TT * NHD * 4);
  _Float16* XSAH  = (_Float16*)alloc((size_t)NTOK * CCH * 2);
  _Float16* XCAH  = (_Float16*)alloc((size_t)NTOK * CCH * 2);
  float*    NORMS = (float*)   alloc(192 * 4);
  float*    GRAM  = (float*)   alloc((size_t)NHD * HDD * HDD * 4);
  float*    ACA   = (float*)   alloc((size_t)NHD * HDD * HDD * 4);
  float*    EPA   = (float*)   alloc((size_t)NTOK * CCH * 4);
  float*    SKIP  = (float*)   alloc((size_t)CCH * NTOK * 4);
  float*    H2F   = (float*)   alloc((size_t)CCH * NTOK * 4);
  _Float16* PADS  = (_Float16*)alloc((size_t)CCH * PSTRC * 2);
  _Float16* PAD1  = (_Float16*)alloc((size_t)CCH * PSTRC * 2);
  _Float16* PAD2  = (_Float16*)alloc((size_t)CCH * PSTRC * 2);
  _Float16* WQH   = (_Float16*)alloc((size_t)384 * CCH * 2);
  _Float16* SRH   = (_Float16*)alloc((size_t)CCH * CCH * 2);
  _Float16* O1H   = (_Float16*)alloc((size_t)48 * CCH * 2);
  _Float16* O2H   = (_Float16*)alloc((size_t)48 * CCH * 2);
  _Float16* C8H   = (_Float16*)alloc((size_t)CCH * CCH * 2);
  _Float16* CW1   = (_Float16*)alloc((size_t)CCH * KC * 2);
  _Float16* CW2   = (_Float16*)alloc((size_t)CCH * KC * 2);
  _Float16* CW3   = (_Float16*)alloc((size_t)CCH * KC * 2);
  _Float16* CW4   = (_Float16*)alloc((size_t)CCH * KC * 2);

  // zero halos / accumulators (capture-safe, deterministic per call)
  hipMemsetAsync(PADS, 0, (size_t)CCH * PSTRC * 2, stream);
  hipMemsetAsync(PAD1, 0, (size_t)CCH * PSTRC * 2, stream);
  hipMemsetAsync(PAD2, 0, (size_t)CCH * PSTRC * 2, stream);
  hipMemsetAsync(GRAM, 0, (size_t)NHD * HDD * HDD * 4, stream);

  auto cvt = [&](const float* s, _Float16* d, int n) {
    k_cvt<<<(n + 255) / 256, 256, 0, stream>>>(s, d, n);
  };
  cvt(w_qkvv, WQH, 384 * CCH);
  cvt(sr_w, SRH, CCH * CCH);
  cvt(o1w, O1H, 48 * CCH);
  cvt(o2w, O2H, 48 * CCH);
  cvt(c8w, C8H, CCH * CCH);
  const int NW = (CCH * KC + 255) / 256;
  k_cvtw<<<NW, 256, 0, stream>>>(c51aw, CW1);
  k_cvtw<<<NW, 256, 0, stream>>>(c51bw, CW2);
  k_cvtw<<<NW, 256, 0, stream>>>(c52aw, CW3);
  k_cvtw<<<NW, 256, 0, stream>>>(c52bw, CW4);

  k_ln<<<(NTOK + 255) / 256, 256, 0, stream>>>(x, norm_w, norm_b, XT, XNH);
  k_gemm<<<NTOK / 16, 256, 0, stream>>>(XNH, WQH, Y, 384, nullptr, 0, 384, 0);
  k_gemm<<<NTOK / 16, 256, 0, stream>>>(XNH, SRH, YSR, CCH, sr_b, 1, CCH, 0);
  k_pool_avg<<<PLP * CCH, 256, 0, stream>>>(YSR, XPP);
  k_pool_lnkv<<<1, 64, 0, stream>>>(XPP, enw, enb, w_kv, KPN, VPOOL);
  k_cpb<<<TT / 256, 256, 0, stream>>>(rct, cpb1w, cpb1b, cpb2w, cpb2b, TAB);
  k_attn<<<NTOK, 128, 0, stream>>>(Y, qe, temp, sls, rbl, lt, lb, KPN, VPOOL, TAB, rpi, XSAH);
  k_colnorm<<<192, 256, 0, stream>>>(Y, NORMS);
  k_gram<<<NHD * 32, 32, 0, stream>>>(Y, GRAM);
  k_casoft<<<1, 96, 0, stream>>>(GRAM, NORMS, temp2, ACA);
  k_xca<<<(NTOK * CCH + 255) / 256, 256, 0, stream>>>(Y, ACA, XCAH);
  k_gemm<<<NTOK / 16, 256, 0, stream>>>(XSAH, O1H, EPA, 48, o1b, 0, CCH, 0);
  k_gemm<<<NTOK / 16, 256, 0, stream>>>(XCAH, O2H, EPA, 48, o2b, 0, CCH, 48);
  k_ax<<<(NTOK * CCH + 255) / 256, 256, 0, stream>>>(XT, gamma, EPA, SKIP, PADS);

  k_conv3<<<200, 256, 0, stream>>>(PADS, CW1, c51abn, nullptr, PAD1, nullptr);
  k_conv3<<<200, 256, 0, stream>>>(PAD1, CW2, c51bbn, SKIP, PAD2, H2F);
  k_conv3<<<200, 256, 0, stream>>>(PAD2, CW3, c52abn, nullptr, PAD1, nullptr);
  k_conv3<<<200, 256, 0, stream>>>(PAD1, CW4, c52bbn, H2F, PAD2, nullptr);
  k_final<<<200, 256, 0, stream>>>(PAD2, C8H, SKIP, c8b, (float*)d_out);
}
